// MemoryUnit_22479858827786
// MI455X (gfx1250) — matrix operand order repaired and checked by reading, not compiled, not run on hardware
//
#include <hip/hip_runtime.h>
#include <cstdint>
#include <cstddef>

// Problem dims (match reference)
#define BD    8
#define SD    2048
#define HD    256
#define MD    4096
#define TOPK  32
#define ROWS  (BD*SD)          // 16384 rows
#define ROWS_PER_BLOCK 16

typedef __attribute__((ext_vector_type(16))) _Float16 v16h;
typedef __attribute__((ext_vector_type(8)))  float    v8f;
typedef __attribute__((ext_vector_type(4)))  float    f4v;
typedef __attribute__((ext_vector_type(4)))  unsigned u4v;

union AFrag { v16h h; unsigned u[8]; };
union BFrag { v16h h; u4v q[2]; };

__device__ __forceinline__ void wave_lds_fence() {
  asm volatile("s_wait_dscnt 0" ::: "memory");
}
__device__ __forceinline__ void wave_store_fence() {
  asm volatile("s_wait_storecnt 0" ::: "memory");
}
// monotonic float->uint key (larger float => larger key)
__device__ __forceinline__ unsigned f2key(float f) {
  unsigned u = __float_as_uint(f);
  return (u & 0x80000000u) ? ~u : (u | 0x80000000u);
}

// ---------------------------------------------------------------------------
// Kernel 0: mn = l2norm(memories) in f16, row-major [M][H].  One wave per row.
// ---------------------------------------------------------------------------
__global__ __launch_bounds__(256)
void memunit_norm_mem(const float* __restrict__ mem, _Float16* __restrict__ mnh) {
  const int wv   = threadIdx.x >> 5;
  const int lane = threadIdx.x & 31;
  const int row  = blockIdx.x * 8 + wv;
  const f4v* src = (const f4v*)(mem + (size_t)row * HD + lane * 8);
  f4v a = src[0], b = src[1];
  float s = a.x*a.x + a.y*a.y + a.z*a.z + a.w*a.w
          + b.x*b.x + b.y*b.y + b.z*b.z + b.w*b.w;
  #pragma unroll
  for (int off = 16; off >= 1; off >>= 1) s += __shfl_xor(s, off, 32);
  const float inv = 1.0f / fmaxf(sqrtf(s), 1e-12f);
  union { _Float16 h[8]; u4v q; } o;
  o.h[0]=(_Float16)(a.x*inv); o.h[1]=(_Float16)(a.y*inv);
  o.h[2]=(_Float16)(a.z*inv); o.h[3]=(_Float16)(a.w*inv);
  o.h[4]=(_Float16)(b.x*inv); o.h[5]=(_Float16)(b.y*inv);
  o.h[6]=(_Float16)(b.z*inv); o.h[7]=(_Float16)(b.w*inv);
  *((u4v*)(mnh + (size_t)row * HD + lane * 8)) = o.q;
}

// ---------------------------------------------------------------------------
// Main kernel: 16 rows per block.  WMMA logits -> LDS, radix top-32, softmax,
// dense weight row + sparse read.
// LDS layout (bytes):
//   logits  : 16*4096*4 = 262144
//   xh      : 16*264*2  =   8448   (row stride padded: conflict-free A loads)
//   norm2   : 16*4, invn: 16*4 (=128)
//   hist    : 8*256*4   =   8192   (per-wave histograms)
//   bcast   : 8*2*4, cnt: 8*4, list: 8*32*4
// total = 280032 bytes  (<= 320KB WGP LDS)
// ---------------------------------------------------------------------------
#define SMEM_BYTES 280032

__global__ __launch_bounds__(256)
void memunit_main(const float* __restrict__ x,
                  const _Float16* __restrict__ mnh,
                  const float* __restrict__ memories,
                  float* __restrict__ readOut,
                  float* __restrict__ weightOut)
{
  extern __shared__ char smem[];
  float*    logits = (float*)smem;                                   // 262144
  _Float16* xh     = (_Float16*)(smem + 262144);                     // 8448
  float*    norm2  = (float*)(smem + 262144 + 8448);                 // 64
  float*    invn   = norm2 + 16;                                     // 64
  unsigned* hist   = (unsigned*)(smem + 262144 + 8448 + 128);        // 8192
  unsigned* bcast  = hist + 8*256;                                   // 64
  unsigned* cnt_   = bcast + 16;                                     // 32
  unsigned* list_  = cnt_ + 8;                                       // 1024

  const int t    = threadIdx.x;
  const int wv   = t >> 5;
  const int lane = t & 31;
  const size_t rowBase = (size_t)blockIdx.x * ROWS_PER_BLOCK;

  // ---------------- Stage A: x tile -> f16 LDS + row norms ----------------
  if (t < 16) norm2[t] = 0.0f;
  __syncthreads();
  {
    const int r  = t >> 4;              // 0..15 row in tile
    const int c0 = (t & 15) * 16;       // 16-col chunk
    const f4v* xr = (const f4v*)(x + (rowBase + r) * HD + c0);
    float ssq = 0.0f;
    #pragma unroll
    for (int j = 0; j < 4; ++j) {
      f4v f = xr[j];
      ssq += f.x*f.x + f.y*f.y + f.z*f.z + f.w*f.w;
      _Float16* xd = xh + r * 264 + c0 + j * 4;
      xd[0]=(_Float16)f.x; xd[1]=(_Float16)f.y;
      xd[2]=(_Float16)f.z; xd[3]=(_Float16)f.w;
    }
    atomicAdd(&norm2[r], ssq);
  }
  __syncthreads();
  if (t < 16) invn[t] = 1.0f / fmaxf(sqrtf(norm2[t]), 1e-12f);

  // ---------------- Stage B: logits via v_wmma_f32_16x16x32_f16 ----------
  const int m  = lane & 15;     // A row / B column within tile
  const int hi = lane >> 4;     // lane half selector
  const unsigned* xh32 = (const unsigned*)xh;

  // A fragments for all 8 K-blocks (ISA 16-bit A 16x32 lane layout)
  AFrag afr[8];
  #pragma unroll
  for (int kb = 0; kb < 8; ++kb) {
    #pragma unroll
    for (int v = 0; v < 8; ++v) {
      const int K0 = kb*32 + ((v < 4) ? (2*v) : (16 + 2*(v-4))) + hi*8;
      afr[kb].u[v] = xh32[(m * 264 + K0) >> 1];   // 2 consecutive halves
    }
  }

  for (int nt = 0; nt < 32; ++nt) {
    const int ntile  = wv * 32 + nt;            // 256 N-tiles of 16 memories
    const int memRow = ntile * 16 + m;          // B column = memory index
    const u4v* bp = (const u4v*)(mnh + (size_t)memRow * HD + hi * 8);
    v8f c = {0.f,0.f,0.f,0.f,0.f,0.f,0.f,0.f};
    #pragma unroll
    for (int kb = 0; kb < 8; ++kb) {
      BFrag bf;
      bf.q[0] = bp[kb*4];
      bf.q[1] = bp[kb*4 + 2];
      c = __builtin_amdgcn_wmma_f32_16x16x32_f16(false, afr[kb].h, false, bf.h,
                                                 (short)0, c, false, false);
    }
    #pragma unroll
    for (int rr = 0; rr < 8; ++rr)
      logits[(rr + 8*hi) * MD + ntile*16 + m] = c[rr];
  }
  __syncthreads();

  // ---------------- Stage C: exact radix top-32 + softmax + outputs -------
  unsigned* histw = hist  + wv * 256;
  unsigned* bcw   = bcast + wv * 2;
  unsigned* cntw  = cnt_  + wv;
  unsigned* listw = list_ + wv * 32;

  for (int rr2 = 0; rr2 < 2; ++rr2) {
    const int row = wv * 2 + rr2;
    const float* lrow = logits + row * MD;
    const f4v* lrow4 = (const f4v*)lrow;

    // cache 128 monotonic keys per lane (col = (i>>2)*128 + lane*4 + (i&3))
    unsigned key[128];
    #pragma unroll
    for (int q = 0; q < 32; ++q) {
      f4v f = lrow4[q * 32 + lane];
      key[q*4+0] = f2key(f.x);
      key[q*4+1] = f2key(f.y);
      key[q*4+2] = f2key(f.z);
      key[q*4+3] = f2key(f.w);
    }

    // 4 rounds of 8-bit radix select -> exact key T of the 32nd largest
    unsigned prefix = 0u, krem = TOPK;
    #pragma unroll
    for (int r = 0; r < 4; ++r) {
      #pragma unroll
      for (int j = 0; j < 8; ++j) histw[lane*8 + j] = 0u;
      wave_lds_fence();
      const int sh = 24 - 8*r;
      #pragma unroll
      for (int i = 0; i < 128; ++i) {
        const unsigned k = key[i];
        bool match;
        if (r == 0) match = true;
        else        match = ((k >> (sh+8)) == (prefix >> (sh+8)));
        if (match) atomicAdd(&histw[(k >> sh) & 255u], 1u);
      }
      wave_lds_fence();
      // lane l covers digits [255-8l-7 .. 255-8l] (descending chunks)
      unsigned h[8]; unsigned s = 0u;
      #pragma unroll
      for (int j = 0; j < 8; ++j) { h[j] = histw[255 - 8*lane - j]; s += h[j]; }
      unsigned c = s;
      #pragma unroll
      for (int off = 1; off < 32; off <<= 1) {
        const unsigned tv = (unsigned)__shfl_up((int)c, off, 32);
        if (lane >= off) c += tv;
      }
      unsigned acc = c - s;   // count of elements with higher digit
      #pragma unroll
      for (int j = 0; j < 8; ++j) {
        const unsigned hv = h[j];
        if (acc < krem && acc + hv >= krem) {   // unique crossing
          bcw[0] = prefix | ((unsigned)(255 - 8*lane - j) << sh);
          bcw[1] = krem - acc;
        }
        acc += hv;
      }
      wave_lds_fence();
      prefix = bcw[0];
      krem   = bcw[1];
    }
    const unsigned T = prefix;

    // collect the 32 winning columns (order-free: softmax is set-invariant)
    if (lane == 0) cntw[0] = 0u;
    wave_lds_fence();
    #pragma unroll
    for (int i = 0; i < 128; ++i) {
      if (key[i] > T) {
        const unsigned p = atomicAdd(cntw, 1u);
        listw[p] = (unsigned)((i >> 2)*128 + (lane << 2) + (i & 3));
      }
    }
    wave_lds_fence();
    #pragma unroll
    for (int i = 0; i < 128; ++i) {
      if (key[i] == T) {
        const unsigned p = atomicAdd(cntw, 1u);
        if (p < 32u) listw[p] = (unsigned)((i >> 2)*128 + (lane << 2) + (i & 3));
      }
    }
    wave_lds_fence();

    const unsigned mycol = listw[lane];
    const float val = lrow[mycol] * invn[row];   // apply 1/||x|| scale

    // softmax across the 32 lanes (temperature = 1)
    float mx = val;
    #pragma unroll
    for (int off = 16; off >= 1; off >>= 1) mx = fmaxf(mx, __shfl_xor(mx, off, 32));
    float e = __expf(val - mx);
    float sum = e;
    #pragma unroll
    for (int off = 16; off >= 1; off >>= 1) sum += __shfl_xor(sum, off, 32);
    const float w = e / sum;

    const size_t grow = rowBase + row;

    // dense weight row: stream zeros then scatter (same-wave store order)
    float* wrow = weightOut + grow * (size_t)MD;
    f4v z = {0.f,0.f,0.f,0.f};
    f4v* wrow4 = (f4v*)wrow;
    #pragma unroll
    for (int it = 0; it < 32; ++it) wrow4[it*32 + lane] = z;
    wave_store_fence();
    wrow[mycol] = w;

    // read = sum_i w_i * memories[col_i]  (raw memories, 8 floats per lane)
    float a0=0,a1=0,a2=0,a3=0,a4=0,a5=0,a6=0,a7=0;
    for (int i = 0; i < 32; ++i) {
      const float wi = __shfl(w, i, 32);
      const int   ci = __shfl((int)mycol, i, 32);
      const f4v* mr = (const f4v*)(memories + (size_t)ci * HD + lane * 8);
      f4v m0 = mr[0], m1 = mr[1];
      a0 += wi*m0.x; a1 += wi*m0.y; a2 += wi*m0.z; a3 += wi*m0.w;
      a4 += wi*m1.x; a5 += wi*m1.y; a6 += wi*m1.z; a7 += wi*m1.w;
    }
    f4v* rr4 = (f4v*)(readOut + grow * (size_t)HD + lane * 8);
    f4v o0 = {a0,a1,a2,a3}, o1 = {a4,a5,a6,a7};
    rr4[0] = o0; rr4[1] = o1;
  }
}

// ---------------------------------------------------------------------------
extern "C" void kernel_launch(void* const* d_in, const int* in_sizes, int n_in,
                              void* d_out, int out_size, void* d_ws, size_t ws_size,
                              hipStream_t stream) {
  (void)in_sizes; (void)n_in; (void)out_size; (void)ws_size;
  const float* x        = (const float*)d_in[0];
  const float* memories = (const float*)d_in[1];
  float* readOut   = (float*)d_out;
  float* weightOut = readOut + (size_t)BD * SD * HD;   // concat: read, weight
  _Float16* mnh    = (_Float16*)d_ws;                  // 2 MB scratch

  hipFuncSetAttribute((const void*)memunit_main,
                      hipFuncAttributeMaxDynamicSharedMemorySize, SMEM_BYTES);

  memunit_norm_mem<<<MD / 8, 256, 0, stream>>>(memories, mnh);
  memunit_main<<<ROWS / ROWS_PER_BLOCK, 256, SMEM_BYTES, stream>>>(
      x, mnh, memories, readOut, weightOut);
}
